// GeometricGlob_Embedding_Original_37855841747290
// MI455X (gfx1250) — matrix operand order repaired and checked by reading, not compiled, not run on hardware
//
#include <hip/hip_runtime.h>
#include <math.h>

#define BATCH 2
#define NPTS 384
#define DMODEL 64
#define KNN 3
#define MTILES (NPTS / 16)         // 24 tiles of 16 m-rows
#define WAVES_PER_BLOCK 4

#define REF_BYTES  (BATCH * NPTS * KNN * 3 * 4)            // 27648, 32B aligned
#define PACK_BYTES (2 * 8 * 32 * 16 * 2)                   // 16384
#define PACK_OFF   REF_BYTES
#define BIAS_OFF   (REF_BYTES + PACK_BYTES)

typedef __attribute__((ext_vector_type(16))) _Float16 v16h;
typedef __attribute__((ext_vector_type(8)))  float    v8f;

__device__ __forceinline__ v8f wmma_f16(v16h a, v16h b, v8f c) {
  // D = A(16x32 f16) * B(32x16 f16) + C(16x16 f32)
  return __builtin_amdgcn_wmma_f32_16x16x32_f16(
      /*neg_a=*/false, a, /*neg_b=*/false, b,
      /*c_mod=*/(short)0, c, /*reuse_a=*/false, /*reuse_b=*/false);
}

// 10000^(-j/32) = 10^(-j/8), j = 0..31 (compile-time folded under full unroll)
__device__ __constant__ const float DT[32] = {
  1.0f, 0.7498942093324559f, 0.5623413251903491f, 0.4216965034285822f,
  0.31622776601683794f, 0.23713737056616552f, 0.17782794100389228f, 0.13335214321633237f,
  0.1f, 0.07498942093324558f, 0.05623413251903491f, 0.04216965034285822f,
  0.031622776601683794f, 0.023713737056616554f, 0.01778279410038923f, 0.013335214321633237f,
  0.01f, 0.007498942093324559f, 0.005623413251903491f, 0.004216965034285822f,
  0.0031622776601683794f, 0.0023713737056616554f, 0.001778279410038923f, 0.0013335214321633236f,
  0.001f, 0.0007498942093324559f, 0.0005623413251903491f, 0.0004216965034285822f,
  0.00031622776601683794f, 0.00023713737056616553f, 0.0001778279410038923f, 0.00013335214321633237f
};

// Build a 16x32 f16 A-fragment slice of the sinusoidal embedding.
// CDNA5 16-bit A layout (16x32): lane half lh picks K subset; VGPR r pair
// (2r,2r+1) is one (sin,cos) pair with frequency index
//   j = slice*16 + 4*lh + (r<4 ? r : r+4).
// `pre` = idx * (1/2pi) * 10^(-lh/2)  (lane-half factor + v_sin revolutions
// scaling pre-folded), so each element is one v_mul + one v_sin/v_cos.
__device__ __forceinline__ v16h build_afrag(float pre, int slice) {
  v16h f;
#pragma unroll
  for (int r = 0; r < 8; ++r) {
    const float dt = DT[slice * 16 + (r < 4 ? r : r + 4)];
    float w = pre * dt;                       // revolutions
    f[2 * r]     = (_Float16)__builtin_amdgcn_sinf(w);
    f[2 * r + 1] = (_Float16)__builtin_amdgcn_cosf(w);
  }
  return f;
}

// Build a 32x16 f16 B-fragment of W^T (used once, in the pack kernel).
// B element (k, col) = W[col][k]; lane l: col = l&15, VGPR r halves hold
// K = 8*(l>>4) + (r<4 ? 2r : 2r+8) + h.
__device__ __forceinline__ v16h build_bfrag(const float* __restrict__ W,
                                            int jtile, int slice, int l) {
  const int lh = l >> 4, col = l & 15;
  const float* rowp = W + (size_t)(jtile * 16 + col) * DMODEL + slice * 32 + 8 * lh;
  v16h f;
#pragma unroll
  for (int r = 0; r < 8; ++r) {
    float2 w = *(const float2*)(rowp + (r < 4 ? 2 * r : 2 * r + 8));
    f[2 * r]     = (_Float16)w.x;
    f[2 * r + 1] = (_Float16)w.y;
  }
  return f;
}

// ---------------------------------------------------------------------------
// Kernel 1: 3-NN search + normalized reference vectors -> ws.ref
// ---------------------------------------------------------------------------
__global__ void ggeo_nn_kernel(const float* __restrict__ xyz,
                               float* __restrict__ ref) {
  int t = blockIdx.x * blockDim.x + threadIdx.x;
  if (t >= BATCH * NPTS) return;
  int b = t / NPTS, n = t % NPTS;
  const float* base = xyz + (size_t)b * NPTS * 3;
  float px = base[n * 3 + 0], py = base[n * 3 + 1], pz = base[n * 3 + 2];
  float d0 = 3.4e38f, d1 = 3.4e38f, d2 = 3.4e38f;
  int i0 = 0, i1 = 0, i2 = 0;
  for (int m = 0; m < NPTS; ++m) {
    if (m == n) continue;
    float dx = base[m * 3 + 0] - px;
    float dy = base[m * 3 + 1] - py;
    float dz = base[m * 3 + 2] - pz;
    float dd = dx * dx + dy * dy + dz * dz;
    if (dd < d0)      { d2 = d1; i2 = i1; d1 = d0; i1 = i0; d0 = dd; i0 = m; }
    else if (dd < d1) { d2 = d1; i2 = i1; d1 = dd; i1 = m; }
    else if (dd < d2) { d2 = dd; i2 = m; }
  }
  int idxs[3] = {i0, i1, i2};
#pragma unroll
  for (int k = 0; k < KNN; ++k) {
    int m = idxs[k];
    float vx = base[m * 3 + 0] - px;
    float vy = base[m * 3 + 1] - py;
    float vz = base[m * 3 + 2] - pz;
    float inv = rsqrtf(vx * vx + vy * vy + vz * vz);
    float* o = ref + ((size_t)(b * NPTS + n) * KNN + k) * 3;
    o[0] = vx * inv; o[1] = vy * inv; o[2] = vz * inv;
  }
}

// ---------------------------------------------------------------------------
// Kernel 2: pack Wd/Wa into lane-strided f16 WMMA B-fragments (once), and
// pre-sum the biases. One block: 16 waves = 2 matrices x 4 jtiles x 2 slices.
// ---------------------------------------------------------------------------
__global__ __launch_bounds__(512)
void ggeo_pack_kernel(const float* __restrict__ Wd, const float* __restrict__ Wa,
                      const float* __restrict__ bd, const float* __restrict__ ba,
                      _Float16* __restrict__ pack, float* __restrict__ biassum) {
  const int t = threadIdx.x;
  const int w = t >> 5, l = t & 31;
  const int mat = w >> 3, frag = w & 7;          // frag = jtile*2 + slice
  const int jtile = frag >> 1, slice = frag & 1;
  const float* W = mat ? Wa : Wd;
  v16h f = build_bfrag(W, jtile, slice, l);
  *(v16h*)(pack + ((size_t)(mat * 8 + frag) * 32 + l) * 16) = f;
  if (t < DMODEL) biassum[t] = bd[t] + ba[t];
}

__device__ __forceinline__ v16h load_bfrag(const _Float16* __restrict__ pack,
                                           int mat, int jtile, int slice, int l) {
  return *(const v16h*)(pack + ((size_t)(mat * 8 + jtile * 2 + slice) * 32 + l) * 16);
}

// ---------------------------------------------------------------------------
// Kernel 3: fused geometry -> sinusoidal embed (direct-to-A-fragment) ->
// WMMA projections -> k-max -> bias -> output. One wave32 per 16-row tile.
// ---------------------------------------------------------------------------
__global__ __launch_bounds__(32 * WAVES_PER_BLOCK)
void ggeo_main_kernel(const float* __restrict__ xyz,
                      const float* __restrict__ ref,
                      const _Float16* __restrict__ pack,
                      const float* __restrict__ biassum,
                      float* __restrict__ out) {
  const int wave = threadIdx.x >> 5;
  const int l    = threadIdx.x & 31;
  const int tile = blockIdx.x * WAVES_PER_BLOCK + wave;
  const int mtile = tile % MTILES;
  const int n     = (tile / MTILES) % NPTS;
  const int b     = tile / (MTILES * NPTS);
  const int lh    = l >> 4;
  const int lane16 = l & 15;

  __builtin_prefetch(pack, 0, 0);                // global_prefetch (L2-hot frags)

  // ---- geometry for this lane's row (lanes l and l+16 share a row but
  // compute disjoint K-subsets of the embedding)
  const float* base = xyz + (size_t)b * NPTS * 3;
  const float px = base[n * 3 + 0], py = base[n * 3 + 1], pz = base[n * 3 + 2];
  const int m = mtile * 16 + lane16;
  const float qx = base[m * 3 + 0], qy = base[m * 3 + 1], qz = base[m * 3 + 2];
  float ax = qx - px, ay = qy - py, az = qz - pz;
  float dd = ax * ax + ay * ay + az * az;
  float dist = __builtin_sqrtf(dd);
  float d_idx = dist * 5.0f;                     // dist / COEF_D
  float inv = rsqrtf(dd);                        // m==n -> NaN (matches ref)
  ax *= inv; ay *= inv; az *= inv;

  // per-lane-half embed premultiplier: 1/(2pi) * 10^(-lh/2)
  const float PRESCALE = lh ? 0.05032921210448796f : 0.15915494309189535f;
  const float FACTOR_A = 180.0f / (15.0f * 3.14159265358979323846f);

  float pre_a[KNN];
#pragma unroll
  for (int k = 0; k < KNN; ++k) {
    const float* rv = ref + ((size_t)(b * NPTS + n) * KNN + k) * 3;
    float rx = rv[0], ry = rv[1], rz = rv[2];
    float cx = ry * az - rz * ay;
    float cy = rz * ax - rx * az;
    float cz = rx * ay - ry * ax;
    float sv = __builtin_sqrtf(cx * cx + cy * cy + cz * cz);
    float cv = rx * ax + ry * ay + rz * az;
    pre_a[k] = atan2f(sv, cv) * (FACTOR_A * 1.0f) * PRESCALE;
  }
  float pre_d = d_idx * PRESCALE;

  // ---- embed directly into WMMA A-fragments (registers only)
  v16h Ad0 = build_afrag(pre_d, 0);
  v16h Ad1 = build_afrag(pre_d, 1);
  v16h Aa[KNN][2];
#pragma unroll
  for (int k = 0; k < KNN; ++k) {
    Aa[k][0] = build_afrag(pre_a[k], 0);
    Aa[k][1] = build_afrag(pre_a[k], 1);
  }

  float* orow = out + (((size_t)b * NPTS + n) * NPTS + (size_t)mtile * 16) * DMODEL;

#pragma unroll
  for (int j = 0; j < 4; ++j) {                  // 4 output-column tiles of 16
    v16h Bd0 = load_bfrag(pack, 0, j, 0, l);
    v16h Bd1 = load_bfrag(pack, 0, j, 1, l);
    v8f accd = {};
    accd = wmma_f16(Ad0, Bd0, accd);
    accd = wmma_f16(Ad1, Bd1, accd);

    v16h Ba0 = load_bfrag(pack, 1, j, 0, l);
    v16h Ba1 = load_bfrag(pack, 1, j, 1, l);
    v8f amax = {};
#pragma unroll
    for (int k = 0; k < KNN; ++k) {
      v8f acc = {};
      acc = wmma_f16(Aa[k][0], Ba0, acc);
      acc = wmma_f16(Aa[k][1], Ba1, acc);
      if (k == 0) {
        amax = acc;
      } else {
#pragma unroll
        for (int r = 0; r < 8; ++r) amax[r] = fmaxf(amax[r], acc[r]);  // -> v_max3
      }
    }

    float biases = biassum[j * 16 + lane16];     // bd+ba, k-uniform so commutes

    // C/D layout: VGPR r -> row M = r + 8*lh, col = lane16 within tile j
#pragma unroll
    for (int r = 0; r < 8; ++r) {
      int mrow = r + 8 * lh;
      orow[(size_t)mrow * DMODEL + j * 16 + lane16] = accd[r] + amax[r] + biases;
    }
  }
}

extern "C" void kernel_launch(void* const* d_in, const int* in_sizes, int n_in,
                              void* d_out, int out_size, void* d_ws, size_t ws_size,
                              hipStream_t stream) {
  const float* xyz = (const float*)d_in[0];  // (2,384,3)
  const float* Wd  = (const float*)d_in[1];  // (64,64)
  const float* bd  = (const float*)d_in[2];  // (64)
  const float* Wa  = (const float*)d_in[3];  // (64,64)
  const float* ba  = (const float*)d_in[4];  // (64)
  float* out = (float*)d_out;                // (2,384,384,64)

  float*    ref     = (float*)d_ws;                              // 27648 B
  _Float16* pack    = (_Float16*)((char*)d_ws + PACK_OFF);       // 16384 B
  float*    biassum = (float*)((char*)d_ws + BIAS_OFF);          // 256 B

  ggeo_nn_kernel<<<(BATCH * NPTS + 255) / 256, 256, 0, stream>>>(xyz, ref);
  ggeo_pack_kernel<<<1, 512, 0, stream>>>(Wd, Wa, bd, ba, pack, biassum);

  const int tiles  = BATCH * NPTS * MTILES;          // 18432 wave-tiles
  const int blocks = tiles / WAVES_PER_BLOCK;        // 4608 blocks of 4 waves
  ggeo_main_kernel<<<blocks, 32 * WAVES_PER_BLOCK, 0, stream>>>(
      xyz, ref, pack, biassum, out);
}
